// VariableSelectionNetwork_77515569758659
// MI455X (gfx1250) — matrix operand order repaired and checked by reading, not compiled, not run on hardware
//
#include <hip/hip_runtime.h>
#include <hip/hip_bf16.h>

// ---------------------------------------------------------------------------
// VariableSelectionNetwork fused CDNA5 (gfx1250) implementation.
//   F=32, H=160, B=64, T=256.  Dominant GEMMs run on v_wmma_f32_16x16x32_f16.
//   Weights are pre-packed into WMMA B-fragment order so every fragment load
//   is one contiguous, fully coalesced 1KB wave transaction.
// ---------------------------------------------------------------------------

typedef __attribute__((ext_vector_type(16))) _Float16 v16h;
typedef __attribute__((ext_vector_type(8)))  _Float16 v8h;
typedef __attribute__((ext_vector_type(8)))  float    v8f;

constexpr int CF = 32;    // F
constexpr int CH = 160;   // H
constexpr int CB = 64;    // B
constexpr int CT = 256;   // T
constexpr int ROWS = 16;  // rows (b,t) per block tile
constexpr int NT = 10;    // N tiles of 16  (H / 16)
constexpr int KT = 5;     // K tiles of 32  (H / 32)
constexpr int STRIDE = 168; // padded LDS row stride (halfwords / floats)
constexpr float LN_EPS = 0.001f;

// fragment-packed panel sizes (halfwords)
constexpr int FRAG_ELEMS = 32 * 16;               // one 32x16 B-fragment: 512 f16
constexpr int PANEL = KT * NT * FRAG_ELEMS;       // one full HxH matrix: 25600 f16

__device__ __forceinline__ v8f wmma_f16(v16h a, v16h b, v8f c) {
    return __builtin_amdgcn_wmma_f32_16x16x32_f16(
        /*neg_a=*/false, a, /*neg_b=*/false, b,
        /*c_mod=*/(short)0, c, /*reuse_a=*/false, /*reuse_b=*/false);
}

__device__ __forceinline__ float elu_f(float z) {
    return z > 0.f ? z : __expf(z) - 1.f;
}
__device__ __forceinline__ float sigmoid_f(float z) {
    return 1.f / (1.f + __expf(-z));
}

// Load one 16x32 f16 A-fragment (ISA 7.12.2 layout) from a row-major LDS
// matrix with row stride STRIDE halfwords.  Two contiguous 16B chunks/lane.
__device__ __forceinline__ v16h load_a_lds(const _Float16* m, int row, int kt, int hl) {
    const _Float16* base = m + row * STRIDE + kt * 32 + hl * 8;
    v8h lo = *(const v8h*)(base);
    v8h hi = *(const v8h*)(base + 16);
    v16h a;
#pragma unroll
    for (int e = 0; e < 8; ++e) { a[e] = lo[e]; a[8 + e] = hi[e]; }
    return a;
}

// Coalesced fragment-packed B load: frag base + lane*16 halfwords (32B/lane,
// 1KB contiguous per wave).
__device__ __forceinline__ v16h load_b_frag(const _Float16* panel, int kt, int nt, int lane) {
    return *(const v16h*)(panel + ((size_t)(kt * NT + nt) * 32 + lane) * 16);
}

// ---------------------------------------------------------------------------
// Kernel 0: cast weights to f16 in WMMA B-fragment-interleaved order.
//   Fragment (kt,nt), lane l, element e holds  W[k][n]  with
//     n = nt*16 + (l&15),  k = kt*32 + (e>>3)*16 + (l>>4)*8 + (e&7)     (ISA 7.12.2 B layout)
//   w2f:  [f][kt][nt][lane][e]   of W2[f][k][n]
//   oXf:  [kt][nt][lane][e]      of oWX[k][n]
// Writes are fully coalesced; reads are scattered but this kernel is tiny.
// ---------------------------------------------------------------------------
__global__ __launch_bounds__(256) void vsn_prep(
    const float* __restrict__ W2, const float* __restrict__ oW1,
    const float* __restrict__ oW2, const float* __restrict__ oWg,
    _Float16* __restrict__ w2f, _Float16* __restrict__ ow1f,
    _Float16* __restrict__ ow2f, _Float16* __restrict__ owgf)
{
    const int W2N = CF * PANEL;       // 819200
    int idx = blockIdx.x * 256 + threadIdx.x;

    if (idx < W2N) {
        int f   = idx / PANEL;
        int rem = idx - f * PANEL;
        int kt  = rem / (NT * FRAG_ELEMS);
        int r2  = rem - kt * (NT * FRAG_ELEMS);
        int nt  = r2 / FRAG_ELEMS;
        int r3  = r2 - nt * FRAG_ELEMS;
        int ln  = r3 >> 4;
        int e   = r3 & 15;
        int n   = nt * 16 + (ln & 15);
        int k   = kt * 32 + (e >> 3) * 16 + (ln >> 4) * 8 + (e & 7);
        w2f[idx] = (_Float16)W2[((size_t)(f * CH + k)) * CH + n];
    } else {
        int j = idx - W2N;
        if (j < 3 * PANEL) {
            int m   = j / PANEL;
            int rem = j - m * PANEL;
            int kt  = rem / (NT * FRAG_ELEMS);
            int r2  = rem - kt * (NT * FRAG_ELEMS);
            int nt  = r2 / FRAG_ELEMS;
            int r3  = r2 - nt * FRAG_ELEMS;
            int ln  = r3 >> 4;
            int e   = r3 & 15;
            int n   = nt * 16 + (ln & 15);
            int k   = kt * 32 + (e >> 3) * 16 + (ln >> 4) * 8 + (e & 7);
            const float* src = (m == 0) ? oW1 : (m == 1) ? oW2 : oWg;
            _Float16*    dst = (m == 0) ? ow1f : (m == 1) ? ow2f : owgf;
            dst[rem] = (_Float16)src[k * CH + n];
        }
    }
}

// ---------------------------------------------------------------------------
// Kernel 1: weights[b,f] = softmax_f( x.reshape(B,T*F) @ Ww + bw )
// One block per b; 256 threads accumulate partial dot products; first wave
// (exactly 32 lanes = wave32) does the softmax with __shfl_xor reductions.
// ---------------------------------------------------------------------------
__global__ __launch_bounds__(256) void vsn_softmax(
    const float* __restrict__ x, const float* __restrict__ Ww,
    const float* __restrict__ bw, float* __restrict__ wts,
    float* __restrict__ w4out)
{
    __shared__ float sacc[CF];
    const int b = blockIdx.x, tid = threadIdx.x;
    if (tid < CF) sacc[tid] = 0.f;
    __syncthreads();

    float part[CF];
#pragma unroll
    for (int f = 0; f < CF; ++f) part[f] = 0.f;

    const int NI = CT * CF;  // 8192
    for (int i = tid; i < NI; i += 256) {
        float xv = x[(size_t)b * NI + i];
        const float* wr = Ww + (size_t)i * CF;
#pragma unroll
        for (int f = 0; f < CF; ++f) part[f] = fmaf(xv, wr[f], part[f]);
    }
#pragma unroll
    for (int f = 0; f < CF; ++f) atomicAdd(&sacc[f], part[f]);
    __syncthreads();

    if (tid < CF) {  // wave 0 only: full wave32, EXEC all ones
        float lg = sacc[tid] + bw[tid];
        float mx = lg;
#pragma unroll
        for (int m = 16; m >= 1; m >>= 1) mx = fmaxf(mx, __shfl_xor(mx, m, 32));
        float e = __expf(lg - mx);
        float sm = e;
#pragma unroll
        for (int m = 16; m >= 1; m >>= 1) sm += __shfl_xor(sm, m, 32);
        float w = e / sm;
        wts[b * CF + tid]   = w;
        w4out[b * CF + tid] = w;  // weights4 output (B,1,1,F)
    }
}

// ---------------------------------------------------------------------------
// Kernel 2: fused main pipeline per 16-row (b,t) tile.
//   4 waves / block.  Each wave owns 8 f-values: builds elu(x*W1+b1) directly
//   into WMMA A fragments, GEMMs vs w2f (L2-resident, fragment-packed), fuses
//   gate + LN + softmax-weighted accumulation into `combined`; waves merge
//   via LDS atomics; then the output GRN (3 WMMA GEMM passes) + final LN.
// ---------------------------------------------------------------------------
__global__ __launch_bounds__(128) void vsn_main(
    const float* __restrict__ x,  const float* __restrict__ W1,
    const float* __restrict__ b1, const float* __restrict__ b2,
    const float* __restrict__ Wg, const float* __restrict__ bg,
    const float* __restrict__ ln_g, const float* __restrict__ ln_b,
    const float* __restrict__ ob1, const float* __restrict__ ob2,
    const float* __restrict__ obg,
    const float* __restrict__ oln_g, const float* __restrict__ oln_b,
    const _Float16* __restrict__ w2f,  const _Float16* __restrict__ ow1f,
    const _Float16* __restrict__ ow2f, const _Float16* __restrict__ owgf,
    const float* __restrict__ wts, float* __restrict__ out)
{
    __shared__ float    xs[ROWS][CF];          // x tile (rows x F)
    __shared__ float    comb[ROWS][STRIDE];    // combined (f32), reused for out-pre-LN
    __shared__ _Float16 combh[ROWS][STRIDE];   // combined as f16 (GEMM A source)
    __shared__ _Float16 t1h[ROWS][STRIDE];     // elu(combined@oW1+ob1) as f16
    __shared__ float    mu[ROWS], rsd[ROWS];

    const int tid = threadIdx.x;
    const int wv  = tid >> 5;          // wave id 0..3 (wave32)
    const int lane = tid & 31;
    const int hl  = lane >> 4;         // half-wave select
    const int l15 = lane & 15;
    const int rowbase = blockIdx.x * ROWS;
    const int bidx = rowbase / CT;     // T divisible by 16 -> one b per tile

    // stage x tile + zero combined accumulator
    for (int i = tid; i < ROWS * CF; i += 128)
        xs[i >> 5][i & 31] = x[(size_t)(rowbase + (i >> 5)) * CF + (i & 31)];
    for (int i = tid; i < ROWS * STRIDE; i += 128)
        (&comb[0][0])[i] = 0.f;
    __syncthreads();

    float cacc[NT][8];
#pragma unroll
    for (int nt = 0; nt < NT; ++nt)
#pragma unroll
        for (int r = 0; r < 8; ++r) cacc[nt][r] = 0.f;

#pragma unroll 1
    for (int fi = 0; fi < 8; ++fi) {
        const int f = wv * 8 + fi;
        const float wf = wts[bidx * CF + f];
        const float xa = xs[l15][f];          // A-layout row scalar
        float xr[8];                          // C-layout row scalars
#pragma unroll
        for (int r = 0; r < 8; ++r) xr[r] = xs[r + 8 * hl][f];

        const _Float16* panel = w2f + (size_t)f * PANEL;
        // prefetch next f's fragment panel (contiguous 51KB) into cache
        if (fi < 7)
            __builtin_prefetch(panel + PANEL, 0, 1);

        // Build A fragments in registers: a = elu(x * W1[f] + b1[f])
        v16h afr[KT];
#pragma unroll
        for (int kt = 0; kt < KT; ++kt) {
#pragma unroll
            for (int e = 0; e < 16; ++e) {
                int k = kt * 32 + (e >> 3) * 16 + hl * 8 + (e & 7);
                float z = fmaf(xa, W1[f * CH + k], b1[f * CH + k]);
                afr[kt][e] = (_Float16)elu_f(z);
            }
        }

        // GEMM vs W2[f] + fused gate; collect row sums for LN
        float feat[NT][8];
        float s[8], sq[8];
#pragma unroll
        for (int r = 0; r < 8; ++r) { s[r] = 0.f; sq[r] = 0.f; }

#pragma unroll
        for (int nt = 0; nt < NT; ++nt) {
            const int n = nt * 16 + l15;
            v8f acc = {};
#pragma unroll
            for (int kt = 0; kt < KT; ++kt) {
                v16h bf = load_b_frag(panel, kt, nt, lane);
                acc = wmma_f16(afr[kt], bf, acc);
            }
            const float b2v = b2[f * CH + n];
            const float wgv = Wg[f * CH + n], bgv = bg[f * CH + n];
#pragma unroll
            for (int r = 0; r < 8; ++r) {
                float av = acc[r] + b2v;
                float g  = sigmoid_f(fmaf(xr[r], wgv, bgv));
                float fv = g * av + (1.f - g) * xr[r];
                feat[nt][r] = fv;
                s[r] += fv; sq[r] += fv * fv;
            }
        }

        // per-row LN stats: rows live in 16-lane halves -> xor-reduce width 16
        float mean[8], rstd[8];
#pragma unroll
        for (int r = 0; r < 8; ++r) {
            float sm = s[r], sv = sq[r];
#pragma unroll
            for (int m = 1; m < 16; m <<= 1) {
                sm += __shfl_xor(sm, m, 16);
                sv += __shfl_xor(sv, m, 16);
            }
            float m_ = sm * (1.f / CH);
            float var = sv * (1.f / CH) - m_ * m_;
            mean[r] = m_;
            rstd[r] = rsqrtf(var + LN_EPS);
        }

        // LN + softmax-weighted accumulation into combined
#pragma unroll
        for (int nt = 0; nt < NT; ++nt) {
            const int n = nt * 16 + l15;
            const float lg = ln_g[f * CH + n], lb = ln_b[f * CH + n];
#pragma unroll
            for (int r = 0; r < 8; ++r)
                cacc[nt][r] += wf * ((feat[nt][r] - mean[r]) * rstd[r] * lg + lb);
        }
    }

    // merge the 4 waves' partial `combined` via LDS f32 atomics (ds_add_f32)
#pragma unroll
    for (int nt = 0; nt < NT; ++nt)
#pragma unroll
        for (int r = 0; r < 8; ++r)
            atomicAdd(&comb[r + 8 * hl][nt * 16 + l15], cacc[nt][r]);
    __syncthreads();

    // f16 copy of combined for GEMM A fragments
    for (int i = tid; i < ROWS * CH; i += 128) {
        int rr = i / CH, hh = i - rr * CH;
        combh[rr][hh] = (_Float16)comb[rr][hh];
    }
    __syncthreads();

    // ---------------- output GRN ----------------
    v16h A1[KT];
#pragma unroll
    for (int kt = 0; kt < KT; ++kt) A1[kt] = load_a_lds(&combh[0][0], l15, kt, hl);

    float og[3][8];
#pragma unroll
    for (int i = 0; i < 3; ++i) {
        const int nt = wv + 4 * i;
        if (nt < NT) {
            const int n = nt * 16 + l15;
            v8f a1 = {}, ag = {};
#pragma unroll
            for (int kt = 0; kt < KT; ++kt) {
                a1 = wmma_f16(A1[kt], load_b_frag(ow1f, kt, nt, lane), a1);
                ag = wmma_f16(A1[kt], load_b_frag(owgf, kt, nt, lane), ag);
            }
            const float o1 = ob1[n], ogb = obg[n];
#pragma unroll
            for (int r = 0; r < 8; ++r) {
                t1h[r + 8 * hl][n] = (_Float16)elu_f(a1[r] + o1);
                og[i][r] = sigmoid_f(ag[r] + ogb);
            }
        }
    }
    __syncthreads();

    v16h A2[KT];
#pragma unroll
    for (int kt = 0; kt < KT; ++kt) A2[kt] = load_a_lds(&t1h[0][0], l15, kt, hl);

#pragma unroll
    for (int i = 0; i < 3; ++i) {
        const int nt = wv + 4 * i;
        if (nt < NT) {
            const int n = nt * 16 + l15;
            v8f a2 = {};
#pragma unroll
            for (int kt = 0; kt < KT; ++kt)
                a2 = wmma_f16(A2[kt], load_b_frag(ow2f, kt, nt, lane), a2);
            const float o2 = ob2[n];
#pragma unroll
            for (int r = 0; r < 8; ++r) {
                float oa = a2[r] + o2;
                float g  = og[i][r];
                float c  = comb[r + 8 * hl][n];
                comb[r + 8 * hl][n] = g * oa + (1.f - g) * c;  // overwrite in place
            }
        }
    }
    __syncthreads();

    // final layernorm over H per row, then store
    if (tid < ROWS) {
        float sm = 0.f, sv = 0.f;
        for (int h = 0; h < CH; ++h) { float v = comb[tid][h]; sm += v; sv += v * v; }
        float m_ = sm * (1.f / CH);
        mu[tid]  = m_;
        rsd[tid] = rsqrtf(sv * (1.f / CH) - m_ * m_ + LN_EPS);
    }
    __syncthreads();

    for (int i = tid; i < ROWS * CH; i += 128) {
        int rr = i / CH, hh = i - rr * CH;
        out[(size_t)(rowbase + rr) * CH + hh] =
            (comb[rr][hh] - mu[rr]) * rsd[rr] * oln_g[hh] + oln_b[hh];
    }
}

// ---------------------------------------------------------------------------
extern "C" void kernel_launch(void* const* d_in, const int* in_sizes, int n_in,
                              void* d_out, int out_size, void* d_ws, size_t ws_size,
                              hipStream_t stream)
{
    const float* x     = (const float*)d_in[0];
    const float* W1    = (const float*)d_in[1];
    const float* b1    = (const float*)d_in[2];
    const float* W2    = (const float*)d_in[3];
    const float* b2    = (const float*)d_in[4];
    const float* Wg    = (const float*)d_in[5];
    const float* bg    = (const float*)d_in[6];
    const float* ln_g  = (const float*)d_in[7];
    const float* ln_b  = (const float*)d_in[8];
    const float* Ww    = (const float*)d_in[9];
    const float* bw    = (const float*)d_in[10];
    const float* oW1   = (const float*)d_in[11];
    const float* ob1   = (const float*)d_in[12];
    const float* oW2   = (const float*)d_in[13];
    const float* ob2   = (const float*)d_in[14];
    const float* oWg   = (const float*)d_in[15];
    const float* obg   = (const float*)d_in[16];
    const float* oln_g = (const float*)d_in[17];
    const float* oln_b = (const float*)d_in[18];

    // workspace layout (~1.8 MB): fragment-packed f16 weights + softmax weights
    char* ws = (char*)d_ws;
    _Float16* w2f  = (_Float16*)ws;                            // F * PANEL
    _Float16* ow1f = (_Float16*)(ws + (size_t)CF * PANEL * 2);
    _Float16* ow2f = ow1f + PANEL;
    _Float16* owgf = ow2f + PANEL;
    float*    wtsp = (float*)(owgf + PANEL);                   // B*F

    float* out = (float*)d_out;
    float* w4  = out + (size_t)CB * CT * CH;  // weights4 tail of d_out

    const int prepN = CF * PANEL + 3 * PANEL;
    vsn_prep<<<(prepN + 255) / 256, 256, 0, stream>>>(
        W2, oW1, oW2, oWg, w2f, ow1f, ow2f, owgf);

    vsn_softmax<<<CB, 256, 0, stream>>>(x, Ww, bw, wtsp, w4);

    vsn_main<<<(CB * CT) / ROWS, 128, 0, stream>>>(
        x, W1, b1, b2, Wg, bg, ln_g, ln_b, ob1, ob2, obg, oln_g, oln_b,
        w2f, ow1f, ow2f, owgf, wtsp, out);
}
